// EntanglementAwarePooling_28381143892394
// MI455X (gfx1250) — matrix operand order repaired and checked by reading, not compiled, not run on hardware
//
#include <hip/hip_runtime.h>
#include <float.h>

#define N_NODES   524288
#define NUM_GRAPHS 8192
#define HDIM       256

#define SCORE_ROWS 128          // rows of x staged per block
#define LDS_STRIDE 264          // 256 + 8 halves pad -> conflict-free ds_load_b128

typedef _Float16 half_t;
typedef __attribute__((ext_vector_type(16))) _Float16 v16h;
typedef __attribute__((ext_vector_type(8)))  _Float16 v8h;
typedef __attribute__((ext_vector_type(8)))  float    v8f;

// ---------------------------------------------------------------------------
// WMMA wrapper: D(f32 16x16) = A(f16 16x32) * B(f16 32x16) + C
// ---------------------------------------------------------------------------
__device__ __forceinline__ v8f wmma_f16(v16h a, v16h b, v8f c) {
  return __builtin_amdgcn_wmma_f32_16x16x32_f16(
      /*neg_a=*/false, a, /*neg_b=*/false, b,
      /*c_mod=*/(short)0, c, /*reuse_a=*/false, /*reuse_b=*/false);
}

// A fragment from two contiguous 16B runs (LDS or global f16).
// Lane layout (ISA 7.12.2): lane<16 -> M=lane, K={0..7,16..23}; lane+16 -> K={8..15,24..31}.
__device__ __forceinline__ v16h load_a_runs(const half_t* p) {
  v8h lo = *(const v8h*)(p + 0);   // K base .. base+7
  v8h hi = *(const v8h*)(p + 16);  // K base+16 .. base+23
  v16h a;
#pragma unroll
  for (int i = 0; i < 8; ++i) { a[i] = lo[i]; a[8 + i] = hi[i]; }
  return a;
}

__device__ __forceinline__ v16h load_a_f16(const half_t* __restrict__ A, int lda,
                                           int row, int k0, int hs) {
  return load_a_runs(A + (size_t)row * lda + k0 + (hs ? 8 : 0));
}

// B fragment from PACKED weights: one 32-byte per-lane load, coalesced.
__device__ __forceinline__ v16h load_b_packed(const half_t* __restrict__ Bp,
                                              int ntiles, int kc, int n0, int lane) {
  const half_t* p =
      Bp + ((size_t)((kc >> 5) * ntiles + (n0 >> 4)) << 9) + (lane << 4);
  return *(const v16h*)p;
}

__device__ __forceinline__ int lower_bound_i32(const int* __restrict__ a, int n, int key) {
  int lo = 0, hi = n;
  while (lo < hi) {
    int mid = (lo + hi) >> 1;
    if (a[mid] < key) lo = mid + 1; else hi = mid;
  }
  return lo;
}

__device__ __forceinline__ float gelu_exact(float v) {
  return 0.5f * v * (1.0f + erff(v * 0.70710678118654752f));
}

// ---------------------------------------------------------------------------
// 0) Pack a row-major f32 [K,N] weight matrix into f16 WMMA-fragment layout.
// ---------------------------------------------------------------------------
__global__ void pack_b_kernel(const float* __restrict__ src,
                              half_t* __restrict__ dst, int K, int N) {
  int idx = blockIdx.x * blockDim.x + threadIdx.x;
  if (idx >= K * N) return;
  int e      = idx & 15;
  int lane   = (idx >> 4) & 31;
  int frag   = idx >> 9;
  int ntiles = N >> 4;
  int ntile  = frag % ntiles;
  int kchunk = frag / ntiles;
  int k = kchunk * 32 + ((e >> 3) << 4) + ((lane >> 4) << 3) + (e & 7);
  int n = ntile * 16 + (lane & 15);
  dst[idx] = (half_t)src[(size_t)k * N + n];
}

// ---------------------------------------------------------------------------
// 1) Attention scores: s = tanh(x @ W1 + b1) @ W2 + b2
//    x tile staged in LDS as f16 (coalesced f32 loads, padded rows), A
//    fragments re-read from LDS per N-tile -> tiny register live set.
// ---------------------------------------------------------------------------
__global__ __launch_bounds__(256) void
score_kernel(const float* __restrict__ x,
             const half_t* __restrict__ W1p,  // packed [256,128]
             const float*  __restrict__ W2,   // [128] f32
             const float*  __restrict__ b1,   // [128]
             const float*  __restrict__ b2,   // [1]
             float* __restrict__ s) {
  extern __shared__ half_t tile[];  // SCORE_ROWS * LDS_STRIDE halves
  const int tid = threadIdx.x;
  const int node_base = blockIdx.x * SCORE_ROWS;

  // Cooperative stage: 128 rows x 256 cols f32 -> f16 LDS (8 floats / thread-iter).
  for (int g = tid; g < SCORE_ROWS * 32; g += 256) {
    int row = g >> 5;
    int c8  = (g & 31) * 8;
    const float* src = x + (size_t)(node_base + row) * HDIM + c8;
    float4 f0 = *(const float4*)(src + 0);
    float4 f1 = *(const float4*)(src + 4);
    v8h h;
    h[0]=(_Float16)f0.x; h[1]=(_Float16)f0.y; h[2]=(_Float16)f0.z; h[3]=(_Float16)f0.w;
    h[4]=(_Float16)f1.x; h[5]=(_Float16)f1.y; h[6]=(_Float16)f1.z; h[7]=(_Float16)f1.w;
    *(v8h*)&tile[row * LDS_STRIDE + c8] = h;
  }
  __syncthreads();

  const int lane = tid & 31;
  const int wv   = tid >> 5;
  const int m    = lane & 15;
  const int hs   = lane >> 4;
  const half_t* arow = &tile[(wv * 16 + m) * LDS_STRIDE + (hs ? 8 : 0)];

  float part[8];
#pragma unroll
  for (int r = 0; r < 8; ++r) part[r] = 0.0f;

  v8f z = {};
#pragma unroll 1
  for (int n = 0; n < 8; ++n) {
    v8f acc0 = z, acc1 = z;   // split dependent WMMA chain in half
#pragma unroll
    for (int c = 0; c < 8; c += 2) {
      v16h a0  = load_a_runs(arow + (c + 0) * 32);
      v16h a1  = load_a_runs(arow + (c + 1) * 32);
      v16h b0  = load_b_packed(W1p, 8, (c + 0) * 32, n * 16, lane);
      v16h b1f = load_b_packed(W1p, 8, (c + 1) * 32, n * 16, lane);
      acc0 = wmma_f16(a0, b0,  acc0);
      acc1 = wmma_f16(a1, b1f, acc1);
    }
    int col = n * 16 + m;
    float b1v = b1[col];
    float w2v = W2[col];
#pragma unroll
    for (int r = 0; r < 8; ++r)
      part[r] += tanhf(acc0[r] + acc1[r] + b1v) * w2v;
  }

#pragma unroll
  for (int off = 8; off >= 1; off >>= 1)
#pragma unroll
    for (int r = 0; r < 8; ++r)
      part[r] += __shfl_xor(part[r], off, 16);

  if (m == 0) {
    float b2v = b2[0];
    int node0 = node_base + wv * 16;
#pragma unroll
    for (int r = 0; r < 8; ++r)
      s[node0 + r + hs * 8] = part[r] + b2v;
  }
}

// ---------------------------------------------------------------------------
// 2) Segment softmax: one wave per graph (batch sorted -> binary search).
// ---------------------------------------------------------------------------
__global__ void softmax_kernel(const float* __restrict__ s,
                               const int* __restrict__ batch,
                               float* __restrict__ w) {
  const int g = blockIdx.x;
  const int lane = threadIdx.x;
  const int start = lower_bound_i32(batch, N_NODES, g);
  const int end   = lower_bound_i32(batch, N_NODES, g + 1);

  float mx = -FLT_MAX;
  for (int i = start + lane; i < end; i += 32) mx = fmaxf(mx, s[i]);
#pragma unroll
  for (int off = 16; off >= 1; off >>= 1) mx = fmaxf(mx, __shfl_xor(mx, off, 32));

  float sum = 0.0f;
  for (int i = start + lane; i < end; i += 32) sum += expf(s[i] - mx);
#pragma unroll
  for (int off = 16; off >= 1; off >>= 1) sum += __shfl_xor(sum, off, 32);

  float inv = (sum > 0.0f) ? 1.0f / sum : 0.0f;
  for (int i = start + lane; i < end; i += 32) w[i] = expf(s[i] - mx) * inv;
}

// ---------------------------------------------------------------------------
// 3) Pooling: one block per graph, thread t owns channel t. No atomics.
// ---------------------------------------------------------------------------
__global__ void pool_kernel(const float* __restrict__ x,
                            const int* __restrict__ batch,
                            const float* __restrict__ w,
                            half_t* __restrict__ pm,
                            half_t* __restrict__ px,
                            half_t* __restrict__ pw) {
  const int g = blockIdx.x;
  const int t = threadIdx.x;
  const int start = lower_bound_i32(batch, N_NODES, g);
  const int end   = lower_bound_i32(batch, N_NODES, g + 1);

  float sum = 0.0f, mx = -FLT_MAX, wsum = 0.0f;
  for (int i = start; i < end; ++i) {
    if (i + 1 < end)
      __builtin_prefetch(x + (size_t)(i + 1) * HDIM + t, 0, 0);
    float xv = x[(size_t)i * HDIM + t];
    float wv = w[i];
    sum  += xv;
    mx    = fmaxf(mx, xv);
    wsum += wv * xv;
  }
  int cnt = end - start;
  float mean = sum / (float)(cnt > 0 ? cnt : 1);
  if (cnt == 0) mx = 0.0f;
  size_t o = (size_t)g * HDIM + t;
  pm[o] = (half_t)mean;
  px[o] = (half_t)mx;
  pw[o] = (half_t)wsum;
}

// ---------------------------------------------------------------------------
// 4) Generic WMMA GEMM: C[row, colOff+col] = act(A@B + bias)
//    Block = 8 waves; wave = 16 rows x 64 cols; grid = (N/64, M/128).
// ---------------------------------------------------------------------------
template <int ACT, bool F16OUT>
__global__ __launch_bounds__(256) void
gemm16_kernel(const half_t* __restrict__ A, int lda,
              const half_t* __restrict__ Bp, int N, int K,
              const float* __restrict__ bias,
              half_t* __restrict__ C16, float* __restrict__ C32,
              int ldc, int colOff) {
  const int lane = threadIdx.x & 31;
  const int wv   = threadIdx.x >> 5;
  const int m    = lane & 15;
  const int hs   = lane >> 4;
  const int row0 = (blockIdx.y * 8 + wv) * 16;
  const int col0 = blockIdx.x * 64;
  const int ntiles = N >> 4;

  v8f acc[4];
  v8f z = {};
#pragma unroll
  for (int n = 0; n < 4; ++n) acc[n] = z;

  for (int kc = 0; kc < K; kc += 32) {
    v16h a = load_a_f16(A, lda, row0 + m, kc, hs);
#pragma unroll
    for (int n = 0; n < 4; ++n) {
      v16h b = load_b_packed(Bp, ntiles, kc, col0 + n * 16, lane);
      acc[n] = wmma_f16(a, b, acc[n]);
    }
  }

#pragma unroll
  for (int n = 0; n < 4; ++n) {
    int col = col0 + n * 16 + m;
    float bv = bias[col];
#pragma unroll
    for (int r = 0; r < 8; ++r) {
      float v = acc[n][r] + bv;
      if (ACT) v = gelu_exact(v);
      int row = row0 + r + hs * 8;
      size_t idx = (size_t)row * ldc + colOff + col;
      if (F16OUT) C16[idx] = (half_t)v;
      else        C32[idx] = v;
    }
  }
}

// ---------------------------------------------------------------------------
// 5) LayerNorm over H=256, in place on d_out. One block per graph row.
// ---------------------------------------------------------------------------
__global__ void ln_kernel(float* __restrict__ out,
                          const float* __restrict__ gamma,
                          const float* __restrict__ beta) {
  const int g = blockIdx.x;
  const int t = threadIdx.x;
  __shared__ float rs[256];
  __shared__ float rq[256];
  float v = out[(size_t)g * HDIM + t];
  rs[t] = v;
  rq[t] = v * v;
  __syncthreads();
  for (int o = 128; o >= 1; o >>= 1) {
    if (t < o) { rs[t] += rs[t + o]; rq[t] += rq[t + o]; }
    __syncthreads();
  }
  float mu  = rs[0] * (1.0f / HDIM);
  float var = rq[0] * (1.0f / HDIM) - mu * mu;
  float rstd = rsqrtf(var + 1e-5f);
  out[(size_t)g * HDIM + t] = (v - mu) * rstd * gamma[t] + beta[t];
}

// ---------------------------------------------------------------------------
// Host launcher
// ---------------------------------------------------------------------------
extern "C" void kernel_launch(void* const* d_in, const int* in_sizes, int n_in,
                              void* d_out, int out_size, void* d_ws, size_t ws_size,
                              hipStream_t stream) {
  const float* x      = (const float*)d_in[0];
  const int*   batch  = (const int*)  d_in[1];
  const float* W_att1 = (const float*)d_in[2];
  const float* b_att1 = (const float*)d_in[3];
  const float* W_att2 = (const float*)d_in[4];
  const float* b_att2 = (const float*)d_in[5];
  const float* Wm  = (const float*)d_in[6];
  const float* bm  = (const float*)d_in[7];
  const float* Wx  = (const float*)d_in[8];
  const float* bx  = (const float*)d_in[9];
  const float* Ww  = (const float*)d_in[10];
  const float* bw  = (const float*)d_in[11];
  const float* Wc1 = (const float*)d_in[12];
  const float* bc1 = (const float*)d_in[13];
  const float* Wc2 = (const float*)d_in[14];
  const float* bc2 = (const float*)d_in[15];
  const float* gamma = (const float*)d_in[16];
  const float* beta  = (const float*)d_in[17];
  float* out = (float*)d_out;

  // Workspace layout
  char* ws = (char*)d_ws;
  size_t off = 0;
  auto alloc = [&](size_t bytes) -> char* {
    char* p = ws + off;
    off += (bytes + 255) & ~(size_t)255;
    return p;
  };
  float*  s      = (float*) alloc((size_t)N_NODES * 4);
  float*  w      = (float*) alloc((size_t)N_NODES * 4);
  half_t* pm     = (half_t*)alloc((size_t)NUM_GRAPHS * HDIM * 2);
  half_t* px     = (half_t*)alloc((size_t)NUM_GRAPHS * HDIM * 2);
  half_t* pw     = (half_t*)alloc((size_t)NUM_GRAPHS * HDIM * 2);
  half_t* comb16 = (half_t*)alloc((size_t)NUM_GRAPHS * 3 * HDIM * 2);
  half_t* c1_16  = (half_t*)alloc((size_t)NUM_GRAPHS * 2 * HDIM * 2);
  half_t* Wa1p   = (half_t*)alloc(256 * 128 * 2);
  half_t* Wmp    = (half_t*)alloc(256 * 256 * 2);
  half_t* Wxp    = (half_t*)alloc(256 * 256 * 2);
  half_t* Wwp    = (half_t*)alloc(256 * 256 * 2);
  half_t* Wc1p   = (half_t*)alloc(768 * 512 * 2);
  half_t* Wc2p   = (half_t*)alloc(512 * 256 * 2);
  (void)ws_size; (void)in_sizes; (void)n_in; (void)out_size;

  auto pack = [&](const float* src, half_t* dst, int K, int N) {
    pack_b_kernel<<<(K * N + 255) / 256, 256, 0, stream>>>(src, dst, K, N);
  };
  pack(W_att1, Wa1p, 256, 128);
  pack(Wm,  Wmp,  256, 256);
  pack(Wx,  Wxp,  256, 256);
  pack(Ww,  Wwp,  256, 256);
  pack(Wc1, Wc1p, 768, 512);
  pack(Wc2, Wc2p, 512, 256);

  // 1) node scores (WMMA, x staged through LDS)
  size_t score_lds = (size_t)SCORE_ROWS * LDS_STRIDE * sizeof(half_t);
  score_kernel<<<N_NODES / SCORE_ROWS, 256, score_lds, stream>>>(
      x, Wa1p, W_att2, b_att1, b_att2, s);

  // 2) segment softmax weights
  softmax_kernel<<<NUM_GRAPHS, 32, 0, stream>>>(s, batch, w);

  // 3) mean/max/weighted pools (streams x once, contention-free)
  pool_kernel<<<NUM_GRAPHS, 256, 0, stream>>>(x, batch, w, pm, px, pw);

  // 4) dense tail: three [G,256]x[256,256] GEMMs into combined [G,768]
  dim3 blk(256);
  gemm16_kernel<0, true><<<dim3(256 / 64, NUM_GRAPHS / 128), blk, 0, stream>>>(
      pm, HDIM, Wmp, 256, 256, bm, comb16, nullptr, 768, 0);
  gemm16_kernel<0, true><<<dim3(256 / 64, NUM_GRAPHS / 128), blk, 0, stream>>>(
      px, HDIM, Wxp, 256, 256, bx, comb16, nullptr, 768, 256);
  gemm16_kernel<0, true><<<dim3(256 / 64, NUM_GRAPHS / 128), blk, 0, stream>>>(
      pw, HDIM, Wwp, 256, 256, bw, comb16, nullptr, 768, 512);

  // [G,768] @ [768,512] + gelu
  gemm16_kernel<1, true><<<dim3(512 / 64, NUM_GRAPHS / 128), blk, 0, stream>>>(
      comb16, 768, Wc1p, 512, 768, bc1, c1_16, nullptr, 512, 0);

  // [G,512] @ [512,256] -> f32 out
  gemm16_kernel<0, false><<<dim3(256 / 64, NUM_GRAPHS / 128), blk, 0, stream>>>(
      c1_16, 512, Wc2p, 256, 512, bc2, nullptr, out, 256, 0);

  // 5) LayerNorm in place
  ln_kernel<<<NUM_GRAPHS, 256, 0, stream>>>(out, gamma, beta);
}
